// MultiHeadAttention_63548336112244
// MI455X (gfx1250) — matrix operand order repaired and checked by reading, not compiled, not run on hardware
//
#include <hip/hip_runtime.h>

#define B_  2
#define S_  2048
#define E_  1024
#define H_  16
#define D_  64
#define BS_ (B_ * S_)  // 4096

typedef unsigned short u16;
typedef __attribute__((ext_vector_type(8)))  u16    u16x8;
typedef __attribute__((ext_vector_type(16))) u16    u16x16;
typedef __attribute__((ext_vector_type(16))) __bf16 v16bf;
typedef __attribute__((ext_vector_type(8)))  float  v8f;

__device__ __forceinline__ u16 f32_to_bf16(float f) {
  unsigned u = __float_as_uint(f);
  u += 0x7FFFu + ((u >> 16) & 1u);  // round-to-nearest-even
  return (u16)(u >> 16);
}

__device__ __forceinline__ v8f zero8() {
  v8f z;
#pragma unroll
  for (int i = 0; i < 8; ++i) z[i] = 0.0f;
  return z;
}

__device__ __forceinline__ v16bf make_frag(u16x8 lo, u16x8 hi) {
  u16x16 v;
#pragma unroll
  for (int i = 0; i < 8; ++i) { v[i] = lo[i]; v[i + 8] = hi[i]; }
  return __builtin_bit_cast(v16bf, v);
}

// A fragment (16x32 bf16): per-lane 16 contiguous halves
__device__ __forceinline__ v16bf frag_a(const u16* p) {
  return make_frag(*reinterpret_cast<const u16x8*>(p),
                   *reinterpret_cast<const u16x8*>(p + 8));
}
// B fragment (32x16 bf16): per-lane 16 contiguous halves
__device__ __forceinline__ v16bf frag_b(const u16* p) {
  return make_frag(*reinterpret_cast<const u16x8*>(p),
                   *reinterpret_cast<const u16x8*>(p + 8));
}

#define WMMA_BF16(a, b, c) \
  __builtin_amdgcn_wmma_f32_16x16x32_bf16(false, (a), false, (b), (short)0, (c), false, false)

// ---------------------------------------------------------------------------
// Streaming fp32 -> bf16 conversion of an activation matrix (row-major copy).
// ---------------------------------------------------------------------------
__global__ __launch_bounds__(256) void conv_act(const float* __restrict__ X,
                                                u16* __restrict__ Y) {
  const size_t i = ((size_t)blockIdx.x * 256 + threadIdx.x) * 8;
  const float4* p = reinterpret_cast<const float4*>(X + i);
  float4 f0 = p[0], f1 = p[1];
  u16x8 o = {f32_to_bf16(f0.x), f32_to_bf16(f0.y), f32_to_bf16(f0.z), f32_to_bf16(f0.w),
             f32_to_bf16(f1.x), f32_to_bf16(f1.y), f32_to_bf16(f1.z), f32_to_bf16(f1.w)};
  *reinterpret_cast<u16x8*>(Y + i) = o;
}

// ---------------------------------------------------------------------------
// Weight convert + transpose: W[k][n] fp32 -> Wt[n][k] bf16 (LDS-tiled 64x64).
// ---------------------------------------------------------------------------
__global__ __launch_bounds__(256) void conv_wt(const float* __restrict__ W,
                                               u16* __restrict__ Wt) {
  __shared__ u16 T[64][72];  // padded stride to avoid bank conflicts
  const int t = threadIdx.x;
  const int k0 = blockIdx.x * 64, n0 = blockIdx.y * 64;
  {
    const int k = t >> 2, nn = (t & 3) * 16;
    const float4* p =
        reinterpret_cast<const float4*>(W + (size_t)(k0 + k) * E_ + n0 + nn);
    float4 f0 = p[0], f1 = p[1], f2 = p[2], f3 = p[3];
    float fv[16] = {f0.x, f0.y, f0.z, f0.w, f1.x, f1.y, f1.z, f1.w,
                    f2.x, f2.y, f2.z, f2.w, f3.x, f3.y, f3.z, f3.w};
#pragma unroll
    for (int i = 0; i < 16; ++i) T[k][nn + i] = f32_to_bf16(fv[i]);
  }
  __syncthreads();
  {
    const int n = t >> 2, kk = (t & 3) * 16;
    u16x8 a, b;
#pragma unroll
    for (int i = 0; i < 8; ++i) { a[i] = T[kk + i][n]; b[i] = T[kk + 8 + i][n]; }
    u16* dst = Wt + (size_t)(n0 + n) * E_ + k0 + kk;
    *reinterpret_cast<u16x8*>(dst)     = a;
    *reinterpret_cast<u16x8*>(dst + 8) = b;
  }
}

// ---------------------------------------------------------------------------
// LDS-free bf16 GEMM: Y = A(bf16,[M,K]) @ Wt(bf16,[N,K]) + bias, bf16 out.
// Block 256 = 8 waves as 2(M)x4(N); wave tile 64x32 = 8 WMMAs / K-step.
// Block tile 128x128; fragments load straight from global (L2-resident).
// vmode 0: out[((b*H+h)*S+s)*D+d]; vmode 1: out[((b*H+h)*D+d)*S+s]
// ---------------------------------------------------------------------------
__global__ __launch_bounds__(256) void proj_gemm(const u16* __restrict__ A,
                                                 const u16* __restrict__ Wt,
                                                 const float* __restrict__ bias,
                                                 u16* __restrict__ out, int vmode) {
  const int tid  = threadIdx.x;
  const int lane = tid & 31, wave = tid >> 5;
  const int hi   = lane >> 4, ln = lane & 15;
  const int r0   = blockIdx.x * 128 + (wave & 1) * 64;
  const int n0   = blockIdx.y * 128 + (wave >> 1) * 32;

  v8f acc[4][2];
#pragma unroll
  for (int mi = 0; mi < 4; ++mi)
#pragma unroll
    for (int ni = 0; ni < 2; ++ni) acc[mi][ni] = zero8();

  for (int kt = 0; kt < E_ / 32; ++kt) {
    const int k0 = kt * 32;
    if (kt + 1 < E_ / 32) {  // prefetch next K tile (global_prefetch_b8)
      __builtin_prefetch(A + (size_t)(r0 + ln) * E_ + k0 + 32, 0, 1);
      __builtin_prefetch(Wt + (size_t)(n0 + ln) * E_ + k0 + 32, 0, 1);
    }
    v16bf af[4], bf[2];
#pragma unroll
    for (int mi = 0; mi < 4; ++mi)
      af[mi] = frag_a(A + (size_t)(r0 + mi * 16 + ln) * E_ + k0 + hi * 16);
#pragma unroll
    for (int ni = 0; ni < 2; ++ni)
      bf[ni] = frag_b(Wt + (size_t)(n0 + ni * 16 + ln) * E_ + k0 + hi * 16);
#pragma unroll
    for (int mi = 0; mi < 4; ++mi)
#pragma unroll
      for (int ni = 0; ni < 2; ++ni)
        acc[mi][ni] = WMMA_BF16(af[mi], bf[ni], acc[mi][ni]);
  }

#pragma unroll
  for (int mi = 0; mi < 4; ++mi)
#pragma unroll
    for (int ni = 0; ni < 2; ++ni)
#pragma unroll
      for (int j = 0; j < 8; ++j) {
        const int row = r0 + mi * 16 + j + 8 * hi;
        const int col = n0 + ni * 16 + ln;
        const u16 bv16 = f32_to_bf16(acc[mi][ni][j] + bias[col]);
        const int b = row >> 11, s = row & (S_ - 1);
        const int h = col >> 6,  d = col & (D_ - 1);
        const size_t idx = (vmode == 0)
            ? ((size_t)(b * H_ + h) * S_ + s) * D_ + d
            : ((size_t)(b * H_ + h) * D_ + d) * S_ + s;
        out[idx] = bv16;
      }
}

// ---------------------------------------------------------------------------
// Causal flash attention, one wave per 16 query rows, key tiles of 32.
// Q,K in [B,H,S,D] bf16; V in [B,H,D,S] bf16. Block 128 = 4 waves.
// ---------------------------------------------------------------------------
__global__ __launch_bounds__(128) void flash_attn(const u16* __restrict__ Qh,
                                                  const u16* __restrict__ Kh,
                                                  const u16* __restrict__ Vt,
                                                  u16* __restrict__ Cc) {
  __shared__ u16 Ps[4][16 * 32];  // wave-private P scratch (C->A relayout)
  const int tid  = threadIdx.x;
  const int lane = tid & 31, wave = tid >> 5;
  const int hi   = lane >> 4, ln = lane & 15;
  const int bh   = blockIdx.y;                 // b*H + h
  const int q0   = blockIdx.x * 64 + wave * 16;

  const u16* Qb = Qh + (size_t)bh * S_ * D_;
  const u16* Kb = Kh + (size_t)bh * S_ * D_;
  const u16* Vb = Vt + (size_t)bh * D_ * S_;
  u16* Pw = Ps[wave];

  const v16bf aq0 = frag_a(Qb + (size_t)(q0 + ln) * D_ + hi * 16);
  const v16bf aq1 = frag_a(Qb + (size_t)(q0 + ln) * D_ + 32 + hi * 16);

  v8f o0 = zero8(), o1 = zero8(), o2 = zero8(), o3 = zero8();
  float m[8], l[8];
#pragma unroll
  for (int j = 0; j < 8; ++j) { m[j] = -3.0e38f; l[j] = 0.0f; }

  const int nkt = (q0 + 47) >> 5;  // causal coverage
  for (int kt = 0; kt < nkt; ++kt) {
    const int k0 = kt * 32;
    if (kt + 1 < nkt) {  // prefetch next key/value tiles
      __builtin_prefetch(Kb + (size_t)(k0 + 32 + ln) * D_, 0, 1);
      __builtin_prefetch(Vb + (size_t)ln * S_ + k0 + 32, 0, 1);
    }
    const u16* klo = Kb + (size_t)(k0 + ln) * D_ + hi * 16;
    const u16* khi = Kb + (size_t)(k0 + 16 + ln) * D_ + hi * 16;
    const v16bf bkl0 = frag_b(klo);
    const v16bf bkl1 = frag_b(klo + 32);
    const v16bf bkh0 = frag_b(khi);
    const v16bf bkh1 = frag_b(khi + 32);

    v8f s0 = zero8(), s1 = zero8();
    s0 = WMMA_BF16(aq0, bkl0, s0);
    s0 = WMMA_BF16(aq1, bkl1, s0);
    s1 = WMMA_BF16(aq0, bkh0, s1);
    s1 = WMMA_BF16(aq1, bkh1, s1);

#pragma unroll
    for (int j = 0; j < 8; ++j) {   // scale 1/sqrt(64) + causal mask
      const int row = q0 + j + 8 * hi;
      const int c0 = k0 + ln, c1 = c0 + 16;
      const float v0 = s0[j] * 0.125f, v1 = s1[j] * 0.125f;
      s0[j] = (c0 <= row) ? v0 : -3.0e38f;
      s1[j] = (c1 <= row) ? v1 : -3.0e38f;
    }
#pragma unroll
    for (int j = 0; j < 8; ++j) {   // online softmax across 16-lane halves
      float rm = fmaxf(s0[j], s1[j]);
      rm = fmaxf(rm, __shfl_xor(rm, 1, 32));
      rm = fmaxf(rm, __shfl_xor(rm, 2, 32));
      rm = fmaxf(rm, __shfl_xor(rm, 4, 32));
      rm = fmaxf(rm, __shfl_xor(rm, 8, 32));
      const float mn   = fmaxf(m[j], rm);
      const float corr = __expf(m[j] - mn);
      const float p0   = __expf(s0[j] - mn);
      const float p1   = __expf(s1[j] - mn);
      float rs = p0 + p1;
      rs += __shfl_xor(rs, 1, 32);
      rs += __shfl_xor(rs, 2, 32);
      rs += __shfl_xor(rs, 4, 32);
      rs += __shfl_xor(rs, 8, 32);
      l[j] = l[j] * corr + rs;
      m[j] = mn;
      o0[j] *= corr; o1[j] *= corr; o2[j] *= corr; o3[j] *= corr;
      Pw[(j + 8 * hi) * 32 + ln]      = f32_to_bf16(p0);
      Pw[(j + 8 * hi) * 32 + 16 + ln] = f32_to_bf16(p1);
    }
    const v16bf ap = frag_a(&Pw[ln * 32 + hi * 16]);  // same-wave LDS in-order
    const v16bf bv0 = frag_b(Vb + (size_t)(0  + ln) * S_ + k0 + hi * 16);
    const v16bf bv1 = frag_b(Vb + (size_t)(16 + ln) * S_ + k0 + hi * 16);
    const v16bf bv2 = frag_b(Vb + (size_t)(32 + ln) * S_ + k0 + hi * 16);
    const v16bf bv3 = frag_b(Vb + (size_t)(48 + ln) * S_ + k0 + hi * 16);
    o0 = WMMA_BF16(ap, bv0, o0);
    o1 = WMMA_BF16(ap, bv1, o1);
    o2 = WMMA_BF16(ap, bv2, o2);
    o3 = WMMA_BF16(ap, bv3, o3);
  }

  const int b = bh >> 4, h = bh & 15;
#pragma unroll
  for (int j = 0; j < 8; ++j) {
    const float inv = 1.0f / l[j];
    const int s = q0 + j + 8 * hi;
    const size_t base = ((size_t)(b * S_ + s)) * E_ + h * D_;
    Cc[base + 0  + ln] = f32_to_bf16(o0[j] * inv);
    Cc[base + 16 + ln] = f32_to_bf16(o1[j] * inv);
    Cc[base + 32 + ln] = f32_to_bf16(o2[j] * inv);
    Cc[base + 48 + ln] = f32_to_bf16(o3[j] * inv);
  }
}

// ---------------------------------------------------------------------------
// LDS-free output GEMM: out(fp32) = Cc(bf16) @ Wt(bf16,[N,K]) + bias.
// ---------------------------------------------------------------------------
__global__ __launch_bounds__(256) void out_gemm(const u16* __restrict__ A,
                                                const u16* __restrict__ Wt,
                                                const float* __restrict__ bias,
                                                float* __restrict__ out) {
  const int tid  = threadIdx.x;
  const int lane = tid & 31, wave = tid >> 5;
  const int hi   = lane >> 4, ln = lane & 15;
  const int r0   = blockIdx.x * 128 + (wave & 1) * 64;
  const int n0   = blockIdx.y * 128 + (wave >> 1) * 32;

  v8f acc[4][2];
#pragma unroll
  for (int mi = 0; mi < 4; ++mi)
#pragma unroll
    for (int ni = 0; ni < 2; ++ni) acc[mi][ni] = zero8();

  for (int kt = 0; kt < E_ / 32; ++kt) {
    const int k0 = kt * 32;
    if (kt + 1 < E_ / 32) {
      __builtin_prefetch(A + (size_t)(r0 + ln) * E_ + k0 + 32, 0, 1);
      __builtin_prefetch(Wt + (size_t)(n0 + ln) * E_ + k0 + 32, 0, 1);
    }
    v16bf af[4], bf[2];
#pragma unroll
    for (int mi = 0; mi < 4; ++mi)
      af[mi] = frag_a(A + (size_t)(r0 + mi * 16 + ln) * E_ + k0 + hi * 16);
#pragma unroll
    for (int ni = 0; ni < 2; ++ni)
      bf[ni] = frag_b(Wt + (size_t)(n0 + ni * 16 + ln) * E_ + k0 + hi * 16);
#pragma unroll
    for (int mi = 0; mi < 4; ++mi)
#pragma unroll
      for (int ni = 0; ni < 2; ++ni)
        acc[mi][ni] = WMMA_BF16(af[mi], bf[ni], acc[mi][ni]);
  }

#pragma unroll
  for (int mi = 0; mi < 4; ++mi)
#pragma unroll
    for (int ni = 0; ni < 2; ++ni)
#pragma unroll
      for (int j = 0; j < 8; ++j) {
        const int row = r0 + mi * 16 + j + 8 * hi;
        const int col = n0 + ni * 16 + ln;
        out[(size_t)row * E_ + col] = acc[mi][ni][j] + bias[col];
      }
}

// ---------------------------------------------------------------------------
extern "C" void kernel_launch(void* const* d_in, const int* in_sizes, int n_in,
                              void* d_out, int out_size, void* d_ws, size_t ws_size,
                              hipStream_t stream) {
  (void)in_sizes; (void)n_in; (void)out_size; (void)ws_size;
  const float* q  = (const float*)d_in[0];
  const float* k  = (const float*)d_in[1];
  const float* v  = (const float*)d_in[2];
  const float* Wq = (const float*)d_in[3];
  const float* bq = (const float*)d_in[4];
  const float* Wk = (const float*)d_in[5];
  const float* bk = (const float*)d_in[6];
  const float* Wv = (const float*)d_in[7];
  const float* bv = (const float*)d_in[8];
  const float* Wo = (const float*)d_in[9];
  const float* bo = (const float*)d_in[10];

  u16* ws = (u16*)d_ws;
  const size_t SEG = (size_t)BS_ * E_;  // 4Mi elems = 8 MB (bf16)
  const size_t WSEG = (size_t)E_ * E_;  // 1Mi elems = 2 MB
  u16* Xc  = ws;                 // activation staging, later aliased as Cc
  u16* Wts = ws + SEG;           // transposed bf16 weight [N,K]
  u16* Qh  = ws + SEG + WSEG;            // [B,H,S,D]
  u16* Kh  = ws + 2 * SEG + WSEG;        // [B,H,S,D]
  u16* Vt  = ws + 3 * SEG + WSEG;        // [B,H,D,S]  (total ws: 34 MB)
  u16* Cc  = Xc;                 // concat [B,S,E], reuses Xc region

  const dim3 gc(BS_ * E_ / (256 * 8));
  const dim3 gw(E_ / 64, E_ / 64);
  const dim3 gp(BS_ / 128, E_ / 128);

  conv_act<<<gc, 256, 0, stream>>>(q, Xc);
  conv_wt <<<gw, 256, 0, stream>>>(Wq, Wts);
  proj_gemm<<<gp, 256, 0, stream>>>(Xc, Wts, bq, Qh, 0);

  conv_act<<<gc, 256, 0, stream>>>(k, Xc);
  conv_wt <<<gw, 256, 0, stream>>>(Wk, Wts);
  proj_gemm<<<gp, 256, 0, stream>>>(Xc, Wts, bk, Kh, 0);

  conv_act<<<gc, 256, 0, stream>>>(v, Xc);
  conv_wt <<<gw, 256, 0, stream>>>(Wv, Wts);
  proj_gemm<<<gp, 256, 0, stream>>>(Xc, Wts, bv, Vt, 1);

  flash_attn<<<dim3(S_ / 64, B_ * H_), 128, 0, stream>>>(Qh, Kh, Vt, Cc);

  conv_wt <<<gw, 256, 0, stream>>>(Wo, Wts);
  out_gemm<<<gp, 256, 0, stream>>>(Cc, Wts, bo, (float*)d_out);
}
